// FnnNormFCIni_13795434955134
// MI455X (gfx1250) — matrix operand order repaired and checked by reading, not compiled, not run on hardware
//
#include <hip/hip_runtime.h>
#include <hip/hip_bf16.h>

#define Bn 4096
#define Cn 128
#define Dn 512
#define Rn 64
#define RG 8            // rules per block (grid.y = Rn/RG = 8)
#define NGRP (Rn / RG)
#define KC 64           // K-chunk per staging step
#define NKC (Dn / KC)   // 8 chunks per rule
#define NG (RG * NKC)   // 64 pipelined chunk iterations per block

#define AStride 520     // 512 + 8 bf16 pad -> 1040 B rows (conflict-free b128)
#define BStride 72      // 64  + 8 bf16 pad ->  144 B rows (conflict-free b128)
#define TILE_ELEMS (Cn * KC)   // one pre-transposed bf16 W tile = 8192 elems

#if defined(__has_builtin)
#if __has_builtin(__builtin_amdgcn_global_load_async_to_lds_b128) && \
    __has_builtin(__builtin_amdgcn_s_wait_asynccnt)
#define USE_ASYNC 1
#endif
#endif

typedef __attribute__((ext_vector_type(16))) __bf16    v16bf;
typedef __attribute__((ext_vector_type(8)))  float     v8f;
typedef __attribute__((ext_vector_type(4)))  unsigned  v4u;
typedef __attribute__((ext_vector_type(2)))  unsigned  v2u;

// async-copy pointer types: 16B int vector in global (AS1) / LDS (AS3)
typedef int v4i __attribute__((vector_size(16)));
typedef v4i __attribute__((address_space(1)))* gptr4;
typedef v4i __attribute__((address_space(3)))* lptr4;

union V16 { v4u q[2]; v16bf v; };

__device__ __forceinline__ unsigned f2bf(float f) {
    unsigned u = __float_as_uint(f);
    unsigned r = u + 0x7FFFu + ((u >> 16) & 1u);   // round-to-nearest-even
    return r >> 16;
}

// ---------------------------------------------------------------- zero init
__global__ void zero_out_kernel(float* __restrict__ out, int n) {
    int i = blockIdx.x * 256 + threadIdx.x;
    if (i < n) out[i] = 0.0f;
}

// ------------------------------------- W pre-transpose + f32->bf16 (once)
// grid = Rn*NKC tiles; tile tb=(r,kc): wt[tb][c][kk] = bf16(W[r][kc*64+kk][c])
__global__ __launch_bounds__(256) void wprep_kernel(
    const float* __restrict__ W, unsigned short* __restrict__ wt) {
    __shared__ float tile[KC * 130];   // 64 x 128 f32, padded rows
    const int tb  = blockIdx.x;
    const int tid = threadIdx.x;
    const float* src = W + (size_t)tb * (KC * Cn);   // tb = r*NKC+kc, K-major
#pragma unroll
    for (int j = 0; j < 32; ++j) {
        int e = tid + 256 * j;
        int kk = e >> 7, c = e & 127;
        tile[kk * 130 + c] = src[kk * Cn + c];       // coalesced reads
    }
    __syncthreads();
    unsigned short* dst = wt + (size_t)tb * TILE_ELEMS;
#pragma unroll
    for (int j = 0; j < 16; ++j) {
        int e2 = tid + 256 * j;          // 4096 bf16-pairs
        int c = e2 >> 5, kp = e2 & 31;
        unsigned lo = f2bf(tile[(kp * 2 + 0) * 130 + c]);
        unsigned hi = f2bf(tile[(kp * 2 + 1) * 130 + c]);
        *(unsigned*)(&dst[c * KC + kp * 2]) = lo | (hi << 16);  // coalesced
    }
}

// ------------------------------------------------- antecedent + softmax(R)
__global__ __launch_bounds__(256) void fire_kernel(
    const float* __restrict__ x, const float* __restrict__ proto,
    const float* __restrict__ var, float* __restrict__ fire) {
    __shared__ float xs[Dn];
    __shared__ float fsL[Rn];
    const int b = blockIdx.x;
    const int t = threadIdx.x;

    xs[t]       = x[b * Dn + t];
    xs[t + 256] = x[b * Dn + 256 + t];
    __syncthreads();

    const int rr = t >> 2;
    const int q  = t & 3;
    const float* pr = proto + rr * Dn + q * 128;
    const float* vr = var   + rr * Dn + q * 128;
    const float* xv = xs + q * 128;
    float s = 0.0f;
#pragma unroll 4
    for (int k = 0; k < 128; ++k) {
        float v  = fminf(fmaxf(vr[k], 1e-4f), 0.1f);
        float d  = xv[k] - pr[k];
        float iv = __frcp_rn(v);
        s += __expf(-(d * d) * (iv * iv));
    }
    s += __shfl_xor(s, 1, 32);
    s += __shfl_xor(s, 2, 32);
    if (q == 0) fsL[rr] = s;
    __syncthreads();

    if (t < Rn) {
        float mx = -3.0e38f;
        for (int i = 0; i < Rn; ++i) mx = fmaxf(mx, fsL[i]);
        float sum = 0.0f;
        for (int i = 0; i < Rn; ++i) sum += __expf(fsL[i] - mx);
        fire[b * Rn + t] = __expf(fsL[t] - mx) / sum;
    }
}

// --------------------------------------- consequent GEMM + ReLU + combine
__global__ __launch_bounds__(256, 1) void expert_kernel(
    const float* __restrict__ x, const unsigned short* __restrict__ wt,
    const float* __restrict__ bias, const float* __restrict__ fire,
    float* __restrict__ out) {
    __shared__ __attribute__((aligned(16))) unsigned short As[64 * AStride];
    __shared__ __attribute__((aligned(16))) unsigned short Bs[2][Cn * BStride];
    __shared__ float fireS[64 * RG];

    const int tid     = threadIdx.x;
    const int lane    = tid & 31;
    const int wid     = tid >> 5;
    const int rowBlk  = wid & 3;
    const int colHalf = wid >> 2;
    const int row0    = blockIdx.x * 64;
    const int rgrp    = blockIdx.y;
    const int hi      = (lane >> 4) & 1;
    const int m       = lane & 15;

    // --- stage x tile (64 x 512) once as bf16
#pragma unroll 8
    for (int j = 0; j < 32; ++j) {
        int e4 = tid + 256 * j;
        int rw = e4 >> 7, k4 = e4 & 127;
        float4 xv = *(const float4*)&x[(row0 + rw) * Dn + k4 * 4];
        v2u val = { f2bf(xv.x) | (f2bf(xv.y) << 16),
                    f2bf(xv.z) | (f2bf(xv.w) << 16) };
        *(v2u*)(&As[rw * AStride + k4 * 4]) = val;
    }
    // --- stage fire-strength sub-block (64 rows x RG rules)
    for (int e = tid; e < 64 * RG; e += 256) {
        int rw = e / RG, rr = e % RG;
        fireS[e] = fire[(row0 + rw) * Rn + rgrp * RG + rr];
    }

    // staging geometry: thread t -> row c = idx>>3, 16B segment seg = idx&7
    const unsigned short* wgrp = wt + (size_t)rgrp * NG * TILE_ELEMS;

#ifndef USE_ASYNC
    v4u pf[4];
#endif

    // --- prologue: stage tile g=0 into buffer 0
    {
        const unsigned short* gt = wgrp;   // tile 0
#ifdef USE_ASYNC
#pragma unroll
        for (int i = 0; i < 4; ++i) {
            int idx = tid + 256 * i;
            int c = idx >> 3, seg = idx & 7;
            __builtin_amdgcn_global_load_async_to_lds_b128(
                (gptr4)(gt + c * KC + seg * 8),
                (lptr4)(&Bs[0][c * BStride + seg * 8]),
                0, 0);
        }
#else
#pragma unroll
        for (int i = 0; i < 4; ++i) {
            int idx = tid + 256 * i;
            int c = idx >> 3, seg = idx & 7;
            pf[i] = *(const v4u*)(gt + c * KC + seg * 8);
        }
#pragma unroll
        for (int i = 0; i < 4; ++i) {
            int idx = tid + 256 * i;
            int c = idx >> 3, seg = idx & 7;
            *(v4u*)(&Bs[0][c * BStride + seg * 8]) = pf[i];
        }
#endif
    }

    v8f persist[4] = {};
    int g = 0;
    for (int rg = 0; rg < RG; ++rg) {
        v8f acc[4] = {};
        for (int kc = 0; kc < NKC; ++kc, ++g) {
            const int buf = g & 1;
#ifdef USE_ASYNC
            __builtin_amdgcn_s_wait_asynccnt(0);   // own async copies landed
#endif
            __syncthreads();   // Bs[buf] staged; Bs[buf^1] readers done

            // --- issue copy of tile g+1 into the other buffer
            const bool more = (g + 1 < NG);
            if (more) {
                const unsigned short* gt = wgrp + (size_t)(g + 1) * TILE_ELEMS;
#ifdef USE_ASYNC
#pragma unroll
                for (int i = 0; i < 4; ++i) {
                    int idx = tid + 256 * i;
                    int c = idx >> 3, seg = idx & 7;
                    __builtin_amdgcn_global_load_async_to_lds_b128(
                        (gptr4)(gt + c * KC + seg * 8),
                        (lptr4)(&Bs[buf ^ 1][c * BStride + seg * 8]),
                        0, 0);
                }
#else
#pragma unroll
                for (int i = 0; i < 4; ++i) {
                    int idx = tid + 256 * i;
                    int c = idx >> 3, seg = idx & 7;
                    pf[i] = *(const v4u*)(gt + c * KC + seg * 8);
                }
#endif
            }

            // --- compute: 8 WMMAs, one-B-fragment-ahead pipeline
            const unsigned short* arow =
                &As[(rowBlk * 16 + m) * AStride + kc * KC + hi * 8];
            V16 av[2];
            av[0].q[0] = *(const v4u*)(arow);
            av[0].q[1] = *(const v4u*)(arow + 16);
            av[1].q[0] = *(const v4u*)(arow + 32);
            av[1].q[1] = *(const v4u*)(arow + 48);

            const unsigned short* bbase =
                &Bs[buf][(colHalf * 64 + m) * BStride + hi * 8];
            V16 bf2[2];
            bf2[0].q[0] = *(const v4u*)(bbase);
            bf2[0].q[1] = *(const v4u*)(bbase + 16);
#pragma unroll
            for (int p = 0; p < 8; ++p) {
                const int ks = p >> 2, t4 = p & 3;
                if (p < 7) {
                    const int pn = p + 1;
                    const unsigned short* bn =
                        bbase + (pn & 3) * (16 * BStride) + (pn >> 2) * 32;
                    bf2[pn & 1].q[0] = *(const v4u*)(bn);
                    bf2[pn & 1].q[1] = *(const v4u*)(bn + 16);
                }
                acc[t4] = __builtin_amdgcn_wmma_f32_16x16x32_bf16(
                    false, av[ks].v, false, bf2[p & 1].v, (short)0,
                    acc[t4], false, false);
            }

#ifndef USE_ASYNC
            if (more) {
#pragma unroll
                for (int i = 0; i < 4; ++i) {
                    int idx = tid + 256 * i;
                    int c = idx >> 3, seg = idx & 7;
                    *(v4u*)(&Bs[buf ^ 1][c * BStride + seg * 8]) = pf[i];
                }
            }
#endif
        }

        // --- rule epilogue: bias + ReLU + fire-weighted accumulation
        const int r = rgrp * RG + rg;
        float fw[8];
#pragma unroll
        for (int j = 0; j < 8; ++j)
            fw[j] = fireS[(rowBlk * 16 + j + 8 * hi) * RG + rg];
#pragma unroll
        for (int t4 = 0; t4 < 4; ++t4) {
            const int cl = colHalf * 64 + t4 * 16 + m;
            const float bb = bias[r * Cn + cl];
#pragma unroll
            for (int j = 0; j < 8; ++j) {
                float e = fmaxf(acc[t4][j] + bb, 0.0f);
                persist[t4][j] += e * fw[j];
            }
        }
    }

    // --- accumulate this rule-group's contribution into out
#pragma unroll
    for (int t4 = 0; t4 < 4; ++t4) {
        const int cl = colHalf * 64 + t4 * 16 + m;
#pragma unroll
        for (int j = 0; j < 8; ++j) {
            const int rw = row0 + rowBlk * 16 + j + 8 * hi;
            atomicAdd(&out[rw * Cn + cl], persist[t4][j]);
        }
    }
}

extern "C" void kernel_launch(void* const* d_in, const int* in_sizes, int n_in,
                              void* d_out, int out_size, void* d_ws, size_t ws_size,
                              hipStream_t stream) {
    const float* x     = (const float*)d_in[0];
    const float* proto = (const float*)d_in[1];
    const float* var   = (const float*)d_in[2];
    const float* W     = (const float*)d_in[3];
    const float* bias  = (const float*)d_in[4];

    float* out  = (float*)d_out;             // [B, C]
    float* fire = out + (size_t)Bn * Cn;     // [B, R]
    unsigned short* wt = (unsigned short*)d_ws;   // [R*NKC][C][KC] bf16, 8.4 MB

    zero_out_kernel<<<(Bn * Cn + 255) / 256, 256, 0, stream>>>(out, Bn * Cn);
    wprep_kernel<<<Rn * NKC, 256, 0, stream>>>(W, wt);
    fire_kernel<<<Bn, 256, 0, stream>>>(x, proto, var, fire);

    dim3 grid(Bn / 64, NGRP);
    expert_kernel<<<grid, 256, 0, stream>>>(x, wt, bias, fire, out);
}
